// STGAT_22849226014867
// MI455X (gfx1250) — matrix operand order repaired and checked by reading, not compiled, not run on hardware
//
#include <hip/hip_runtime.h>

// Problem constants (from the reference)
#define Nn 20000
#define Ee 640000
#define Wn 8
#define Fn 128
#define NE_TOT (Ee + Nn)   // edges + self loops

typedef __attribute__((ext_vector_type(16))) _Float16 v16h;
typedef __attribute__((ext_vector_type(8)))  float    v8f;
typedef int vsi4 __attribute__((vector_size(16)));   // matches async-LDS builtin param

#define FPITCH 36          // LDS row pitch in floats (32 + pad)
#define ORD_NEG_INF 0x007FFFFFu   // f2ord(-inf)

#define GLOBAL_AS __attribute__((address_space(1)))
#define LDS_AS    __attribute__((address_space(3)))

#ifndef __has_builtin
#define __has_builtin(x) 0
#endif

#if __has_builtin(__builtin_amdgcn_global_load_async_to_lds_b128)
#define HAVE_ASYNC_LDS 1
#else
#define HAVE_ASYNC_LDS 0
#endif

#if __has_builtin(__builtin_amdgcn_s_wait_asynccnt)
#define WAIT_ASYNC() __builtin_amdgcn_s_wait_asynccnt(0)
#else
#define WAIT_ASYNC() asm volatile("s_wait_asynccnt 0x0" ::: "memory")
#endif

#if HAVE_ASYNC_LDS
// copy 16B global -> LDS, tracked by ASYNCcnt; imm offset applies to both sides
#define ASYNC_CP16(gp, lp, off)                                          \
  __builtin_amdgcn_global_load_async_to_lds_b128(                        \
      (GLOBAL_AS vsi4*)(gp), (LDS_AS vsi4*)(lp), (off), 0)
#endif

__device__ __forceinline__ unsigned f2ord(float f) {
  unsigned u = __float_as_uint(f);
  return (u & 0x80000000u) ? ~u : (u | 0x80000000u);
}
__device__ __forceinline__ float ord2f(unsigned u) {
  return (u & 0x80000000u) ? __uint_as_float(u & 0x7FFFFFFFu) : __uint_as_float(~u);
}
__device__ __forceinline__ float leaky(float e) { return e > 0.f ? e : 0.2f * e; }

// ----------------------------------------------------------------------------
// WMMA GEMM: C[M x Ncols](f32) = A[M x K](f32, row stride lda) * BT (f16,
// column-major: BT[col*K + k]).  Block = 256 threads = 8 waves; each wave owns
// a 16-row x 64-col tile (4 f32 16x16 accumulators). grid.y tiles columns.
// A tiles are staged f32 into LDS with async-to-LDS copies (ASYNCcnt path),
// double-buffered so WMMA issue overlaps the next tile fill.
// ----------------------------------------------------------------------------
__global__ __launch_bounds__(256) void gemm_wmma_kernel(
    const float* __restrict__ A, int lda,
    const _Float16* __restrict__ BT, int K,
    float* __restrict__ C, int ldc, int M)
{
  __shared__ float ldsA[2][128 * FPITCH];

  const int tid  = threadIdx.x;
  const int lane = tid & 31;
  const int wv   = tid >> 5;
  const int rowTile = blockIdx.x * 128;
  const int c0      = blockIdx.y * 64;
  const int srow = tid >> 1;             // staging: thread -> (row, 16-float half)
  const int scol = (tid & 1) * 16;
  const int arow = rowTile + srow;
  const bool rowOK = (arow < M);
  const int hsel = (lane < 16) ? 0 : 1;  // lane-half select
  const int mloc = wv * 16 + (lane & 15);

  v8f acc[4] = {};

  __builtin_amdgcn_s_wait_tensorcnt(0);  // CDNA5 split wait counter (no-op safe)

  // zero-fill guard rows once (both buffers); async path never writes them
  if (!rowOK) {
#pragma unroll
    for (int i = 0; i < 16; i++) {
      ldsA[0][srow * FPITCH + scol + i] = 0.f;
      ldsA[1][srow * FPITCH + scol + i] = 0.f;
    }
  }

  float* gsrc = const_cast<float*>(A) + (size_t)arow * lda + scol;   // + k0 per step

  // ---- stage k0 = 0 into buffer 0
#if HAVE_ASYNC_LDS
  if (rowOK) {
    float* lp = &ldsA[0][srow * FPITCH + scol];
    ASYNC_CP16(gsrc, lp, 0);
    ASYNC_CP16(gsrc, lp, 16);
    ASYNC_CP16(gsrc, lp, 32);
    ASYNC_CP16(gsrc, lp, 48);
  }
  WAIT_ASYNC();
#else
  if (rowOK) {
    const float4* sp = reinterpret_cast<const float4*>(gsrc);
    float4* dp = reinterpret_cast<float4*>(&ldsA[0][srow * FPITCH + scol]);
    dp[0] = sp[0]; dp[1] = sp[1]; dp[2] = sp[2]; dp[3] = sp[3];
  }
#endif
  __syncthreads();

  int buf = 0;
  for (int k0 = 0; k0 < K; k0 += 32) {
    const int k1 = k0 + 32;

    // ---- kick off async fill of the other buffer for the next K-step
    if (k1 < K) {
#if HAVE_ASYNC_LDS
      if (rowOK) {
        float* lp = &ldsA[buf ^ 1][srow * FPITCH + scol];
        float* gp = gsrc + k1;
        ASYNC_CP16(gp, lp, 0);
        ASYNC_CP16(gp, lp, 16);
        ASYNC_CP16(gp, lp, 32);
        ASYNC_CP16(gp, lp, 48);
      }
#else
      if (rowOK) __builtin_prefetch(gsrc + k1, 0, 1);   // global_prefetch_b8
#endif
    }

    // ---- A fragment (contiguous k = 16*hsel + i, same order as B), f32 LDS -> f16 regs:
    // lanes 0-15: row=mloc, K = 0..15; lanes 16-31: K = 16..31
    const float* ap = &ldsA[buf][mloc * FPITCH + hsel * 16];
    v16h a;
#pragma unroll
    for (int i = 0; i < 8; i++) {
      a[i]     = (_Float16)ap[i];
      a[8 + i] = (_Float16)ap[8 + i];
    }

    // ---- B fragments + WMMA. Lane holds column N=lane&15;
    // lanes 0-15 K=kb..kb+15, lanes 16-31 K=kb+16..kb+31 (via hsel).
    const int kb = k0 + hsel * 16;
#pragma unroll
    for (int j = 0; j < 4; j++) {
      const int col = c0 + j * 16 + (lane & 15);
      const _Float16* bp = BT + (size_t)col * K + kb;
      v16h b;
#pragma unroll
      for (int i = 0; i < 16; i++) b[i] = bp[i];
      acc[j] = __builtin_amdgcn_wmma_f32_16x16x32_f16(
          /*neg_a=*/false, a, /*neg_b=*/false, b,
          /*c_mod=*/(short)0, acc[j], /*reuse_a=*/false, /*reuse_b=*/false);
    }

    if (k1 < K) {
#if HAVE_ASYNC_LDS
      WAIT_ASYNC();          // my async fills of buf^1 landed
      __syncthreads();       // everyone's fills visible; all reads of buf done
#else
      __syncthreads();       // all reads of buf done before we overwrite buf^1
      if (rowOK) {
        const float4* sp = reinterpret_cast<const float4*>(gsrc + k1);
        float4* dp = reinterpret_cast<float4*>(&ldsA[buf ^ 1][srow * FPITCH + scol]);
        dp[0] = sp[0]; dp[1] = sp[1]; dp[2] = sp[2]; dp[3] = sp[3];
      }
      __syncthreads();
#endif
      buf ^= 1;
    }
  }

  // ---- epilogue: C/D layout: VGPR r -> (M = r + hsel*8, N = lane&15)
#pragma unroll
  for (int j = 0; j < 4; j++) {
    const int col = c0 + j * 16 + (lane & 15);
#pragma unroll
    for (int r = 0; r < 8; r++) {
      const int row = rowTile + wv * 16 + r + hsel * 8;
      if (row < M) C[(size_t)row * ldc + col] = acc[j][r];
    }
  }
}

// ----------------------------------------------------------------------------
// prep: w1 [128,256] -> w1T f16 [256][128]; w2 [256,64] -> w2T f16 [64][256];
// zero temporal accumulator.
// ----------------------------------------------------------------------------
__global__ void prep_kernel(const float* __restrict__ w1, const float* __restrict__ w2,
                            _Float16* __restrict__ w1T, _Float16* __restrict__ w2T,
                            float* __restrict__ h_acc)
{
  const int idx = blockIdx.x * blockDim.x + threadIdx.x;
  if (idx < Fn * 256) { int c = idx / Fn, k = idx % Fn; w1T[idx] = (_Float16)w1[k * 256 + c]; }
  if (idx < 256 * 64) { int c = idx / 256, k = idx % 256; w2T[idx] = (_Float16)w2[k * 64 + c]; }
  if (idx < Nn * 64)  h_acc[idx] = 0.f;
}

// per-window reset of aggregation buffers / softmax state
__global__ void init_window_kernel(float* __restrict__ out1, unsigned* __restrict__ m1,
                                   float* __restrict__ z1, float* __restrict__ out2,
                                   unsigned* __restrict__ m2, float* __restrict__ z2)
{
  const int idx = blockIdx.x * blockDim.x + threadIdx.x;
  if (idx < Nn * 256) out1[idx] = 0.f;
  if (idx < Nn * 4)   { m1[idx] = ORD_NEG_INF; z1[idx] = 0.f; }
  if (idx < Nn * 64)  out2[idx] = 0.f;
  if (idx < Nn)       { m2[idx] = ORD_NEG_INF; z2[idx] = 0.f; }
}

// ----------------------------------------------------------------------------
// attention scores: e_s[n,h] = <h[n,h,:], a_src[h,:]>, likewise e_d. Wave/node-head.
// ----------------------------------------------------------------------------
__global__ __launch_bounds__(256) void att_scores_kernel(
    const float* __restrict__ h, const float* __restrict__ a_src,
    const float* __restrict__ a_dst, float* __restrict__ e_s, float* __restrict__ e_d,
    int total, int H)
{
  const int wid  = (blockIdx.x * 256 + threadIdx.x) >> 5;
  const int lane = threadIdx.x & 31;
  if (wid >= total) return;
  const int hh = wid % H;
  const float* hp = h + (size_t)wid * 64;            // [n*H + h] row of 64
  const float v0 = hp[lane], v1 = hp[lane + 32];
  float s = v0 * a_src[hh * 64 + lane] + v1 * a_src[hh * 64 + lane + 32];
  float d = v0 * a_dst[hh * 64 + lane] + v1 * a_dst[hh * 64 + lane + 32];
#pragma unroll
  for (int m = 16; m; m >>= 1) { s += __shfl_xor(s, m, 32); d += __shfl_xor(d, m, 32); }
  if (lane == 0) { e_s[wid] = s; e_d[wid] = d; }
}

// segment max over dst (ordered-uint float atomicMax)
__global__ void edge_max_kernel(const int* __restrict__ ei, const float* __restrict__ e_s,
                                const float* __restrict__ e_d, unsigned* __restrict__ mbuf,
                                int H, int nTot)
{
  const int idx = blockIdx.x * blockDim.x + threadIdx.x;
  if (idx >= nTot) return;
  const int eid = idx / H, hh = idx - eid * H;
  int src, dst;
  if (eid < Ee) { src = ei[eid]; dst = ei[Ee + eid]; } else { src = dst = eid - Ee; }
  const float e = leaky(e_s[src * H + hh] + e_d[dst * H + hh]);
  atomicMax(&mbuf[dst * H + hh], f2ord(e));
}

// segment sum of exp(e - m)
__global__ void edge_sum_kernel(const int* __restrict__ ei, const float* __restrict__ e_s,
                                const float* __restrict__ e_d, const unsigned* __restrict__ mbuf,
                                float* __restrict__ zbuf, int H, int nTot)
{
  const int idx = blockIdx.x * blockDim.x + threadIdx.x;
  if (idx >= nTot) return;
  const int eid = idx / H, hh = idx - eid * H;
  int src, dst;
  if (eid < Ee) { src = ei[eid]; dst = ei[Ee + eid]; } else { src = dst = eid - Ee; }
  const float e = leaky(e_s[src * H + hh] + e_d[dst * H + hh]);
  atomicAdd(&zbuf[dst * H + hh], __expf(e - ord2f(mbuf[dst * H + hh])));
}

// scatter: out[dst, c] += alpha * h[src, c]; 256/(H*D) edges per block, coalesced
__global__ __launch_bounds__(256) void edge_scatter_kernel(
    const int* __restrict__ ei, const float* __restrict__ e_s, const float* __restrict__ e_d,
    const unsigned* __restrict__ mbuf, const float* __restrict__ zbuf,
    const float* __restrict__ hsrc, float* __restrict__ outp, int H, int D)
{
  const int C = H * D;
  const int epb = 256 / C;
  const int local = threadIdx.x / C;
  const int ch = threadIdx.x - local * C;
  const int eid = blockIdx.x * epb + local;
  if (eid >= NE_TOT) return;
  const int hh = ch / D;
  int src, dst;
  if (eid < Ee) { src = ei[eid]; dst = ei[Ee + eid]; } else { src = dst = eid - Ee; }
  const float e = leaky(e_s[src * H + hh] + e_d[dst * H + hh]);
  const float alpha = __expf(e - ord2f(mbuf[dst * H + hh])) / (zbuf[dst * H + hh] + 1e-16f);
  atomicAdd(&outp[(size_t)dst * C + ch], alpha * hsrc[(size_t)src * C + ch]);
}

// bias + relu (layer-1 epilogue)
__global__ void bias_relu_kernel(const float* __restrict__ in, const float* __restrict__ bias,
                                 float* __restrict__ outp, int C, int total)
{
  const int idx = blockIdx.x * blockDim.x + threadIdx.x;
  if (idx >= total) return;
  const float v = in[idx] + bias[idx % C];
  outp[idx] = v > 0.f ? v : 0.f;
}

// layer-2 epilogue: bias + relu + LayerNorm(64), accumulate temporal mean, keep last
__global__ __launch_bounds__(256) void ln_accum_kernel(
    const float* __restrict__ out2, const float* __restrict__ b2,
    const float* __restrict__ gamma, const float* __restrict__ beta,
    float* __restrict__ h_acc, float* __restrict__ h_last, int isLast)
{
  const int wid  = (blockIdx.x * 256 + threadIdx.x) >> 5;
  const int lane = threadIdx.x & 31;
  if (wid >= Nn) return;
  const float* p = out2 + (size_t)wid * 64;
  float v0 = fmaxf(p[lane] + b2[lane], 0.f);
  float v1 = fmaxf(p[lane + 32] + b2[lane + 32], 0.f);
  float s = v0 + v1;
#pragma unroll
  for (int m = 16; m; m >>= 1) s += __shfl_xor(s, m, 32);
  const float mu = s * (1.f / 64.f);
  const float d0 = v0 - mu, d1 = v1 - mu;
  float q = d0 * d0 + d1 * d1;
#pragma unroll
  for (int m = 16; m; m >>= 1) q += __shfl_xor(q, m, 32);
  const float inv = rsqrtf(q * (1.f / 64.f) + 1e-5f);
  const float r0 = d0 * inv * gamma[lane] + beta[lane];
  const float r1 = d1 * inv * gamma[lane + 32] + beta[lane + 32];
  h_acc[(size_t)wid * 64 + lane]      += r0;
  h_acc[(size_t)wid * 64 + lane + 32] += r1;
  if (isLast) {
    h_last[(size_t)wid * 64 + lane]      = r0;
    h_last[(size_t)wid * 64 + lane + 32] = r1;
  }
}

// final: out[n] = <h_last + 0.3*mean_w(h_acc), fc_w> + fc_b
__global__ __launch_bounds__(256) void final_kernel(
    const float* __restrict__ h_last, const float* __restrict__ h_acc,
    const float* __restrict__ fc_w, const float* __restrict__ fc_b,
    float* __restrict__ outp)
{
  const int wid  = (blockIdx.x * 256 + threadIdx.x) >> 5;
  const int lane = threadIdx.x & 31;
  if (wid >= Nn) return;
  const size_t base = (size_t)wid * 64;
  const float c0 = h_last[base + lane]      + 0.3f * h_acc[base + lane]      * (1.f / (float)Wn);
  const float c1 = h_last[base + lane + 32] + 0.3f * h_acc[base + lane + 32] * (1.f / (float)Wn);
  float s = c0 * fc_w[lane] + c1 * fc_w[lane + 32];
#pragma unroll
  for (int m = 16; m; m >>= 1) s += __shfl_xor(s, m, 32);
  if (lane == 0) outp[wid] = s + fc_b[0];
}

// ----------------------------------------------------------------------------
extern "C" void kernel_launch(void* const* d_in, const int* in_sizes, int n_in,
                              void* d_out, int out_size, void* d_ws, size_t ws_size,
                              hipStream_t stream) {
  (void)in_sizes; (void)n_in; (void)out_size; (void)ws_size;
  const float* x   = (const float*)d_in[0];
  const int*   ei  = (const int*)d_in[1];
  const float* w1  = (const float*)d_in[2];
  const float* a1s = (const float*)d_in[3];
  const float* a1d = (const float*)d_in[4];
  const float* b1  = (const float*)d_in[5];
  const float* w2  = (const float*)d_in[6];
  const float* a2s = (const float*)d_in[7];
  const float* a2d = (const float*)d_in[8];
  const float* b2  = (const float*)d_in[9];
  const float* lng = (const float*)d_in[10];
  const float* lnb = (const float*)d_in[11];
  const float* fcw = (const float*)d_in[12];
  const float* fcb = (const float*)d_in[13];
  float* outp = (float*)d_out;

  // carve workspace
  char* ws = (char*)d_ws;
  auto carve = [&](size_t bytes) { void* p = (void*)ws; ws += (bytes + 255) & ~(size_t)255; return p; };
  float*    OUT1  = (float*)carve((size_t)Nn * 256 * 4);
  float*    H1    = (float*)carve((size_t)Nn * 256 * 4);   // GEMM1 out, then relu'd h
  float*    ES1   = (float*)carve((size_t)Nn * 4 * 4);
  float*    ED1   = (float*)carve((size_t)Nn * 4 * 4);
  unsigned* M1    = (unsigned*)carve((size_t)Nn * 4 * 4);
  float*    Z1    = (float*)carve((size_t)Nn * 4 * 4);
  float*    H2R   = (float*)carve((size_t)Nn * 64 * 4);
  float*    OUT2  = (float*)carve((size_t)Nn * 64 * 4);
  float*    ES2   = (float*)carve((size_t)Nn * 4);
  float*    ED2   = (float*)carve((size_t)Nn * 4);
  unsigned* M2    = (unsigned*)carve((size_t)Nn * 4);
  float*    Z2    = (float*)carve((size_t)Nn * 4);
  float*    HACC  = (float*)carve((size_t)Nn * 64 * 4);
  float*    HLAST = (float*)carve((size_t)Nn * 64 * 4);
  _Float16* W1T   = (_Float16*)carve((size_t)256 * Fn * 2);
  _Float16* W2T   = (_Float16*)carve((size_t)64 * 256 * 2);

  prep_kernel<<<(Nn * 64 + 255) / 256, 256, 0, stream>>>(w1, w2, W1T, W2T, HACC);

  const int gemmBlocksM = (Nn + 127) / 128;  // 157
  for (int w = 0; w < Wn; w++) {
    init_window_kernel<<<(Nn * 256 + 255) / 256, 256, 0, stream>>>(OUT1, M1, Z1, OUT2, M2, Z2);

    // ---- GAT layer 1 ----
    gemm_wmma_kernel<<<dim3(gemmBlocksM, 4), 256, 0, stream>>>(
        x + (size_t)w * Fn, Wn * Fn, W1T, Fn, H1, 256, Nn);
    att_scores_kernel<<<(Nn * 4 + 7) / 8, 256, 0, stream>>>(H1, a1s, a1d, ES1, ED1, Nn * 4, 4);
    edge_max_kernel<<<(NE_TOT * 4 + 255) / 256, 256, 0, stream>>>(ei, ES1, ED1, M1, 4, NE_TOT * 4);
    edge_sum_kernel<<<(NE_TOT * 4 + 255) / 256, 256, 0, stream>>>(ei, ES1, ED1, M1, Z1, 4, NE_TOT * 4);
    edge_scatter_kernel<<<NE_TOT, 256, 0, stream>>>(ei, ES1, ED1, M1, Z1, H1, OUT1, 4, 64);
    bias_relu_kernel<<<(Nn * 256 + 255) / 256, 256, 0, stream>>>(OUT1, b1, H1, 256, Nn * 256);

    // ---- GAT layer 2 ----
    gemm_wmma_kernel<<<dim3(gemmBlocksM, 1), 256, 0, stream>>>(H1, 256, W2T, 256, H2R, 64, Nn);
    att_scores_kernel<<<(Nn + 7) / 8, 256, 0, stream>>>(H2R, a2s, a2d, ES2, ED2, Nn, 1);
    edge_max_kernel<<<(NE_TOT + 255) / 256, 256, 0, stream>>>(ei, ES2, ED2, M2, 1, NE_TOT);
    edge_sum_kernel<<<(NE_TOT + 255) / 256, 256, 0, stream>>>(ei, ES2, ED2, M2, Z2, 1, NE_TOT);
    edge_scatter_kernel<<<(NE_TOT + 3) / 4, 256, 0, stream>>>(ei, ES2, ED2, M2, Z2, H2R, OUT2, 1, 64);

    ln_accum_kernel<<<(Nn + 7) / 8, 256, 0, stream>>>(OUT2, b2, lng, lnb, HACC, HLAST, (w == Wn - 1) ? 1 : 0);
  }

  final_kernel<<<(Nn + 7) / 8, 256, 0, stream>>>(HLAST, HACC, fcw, fcb, outp);
}
